// ReactionGAT_90890097918028
// MI455X (gfx1250) — compile-verified, matrix operand order checked by reading, not run on hardware
//
#include <hip/hip_runtime.h>
#include <hip/hip_bf16.h>
#include <math.h>

typedef __bf16 bf16;
typedef __attribute__((ext_vector_type(16))) __bf16 bf16x16;
typedef __attribute__((ext_vector_type(8)))  float  f32x8;

#define NN  32000
#define EE  512000
#define DD  128
#define HHD 4
#define EAA 16
#define ETOT (EE + NN)   // edges + self loops
#define KP1 288          // 2*128 + 16 padded up to multiple of 32

// ---------------- WMMA helpers ----------------

__device__ __forceinline__ bf16x16 load_frag(const bf16* p) {
  // p points at element (row_of_lane, k0 + (lane>>4)*8); 16-bit A/B 16x32 layout:
  // lane half 0: K = [0..7] and [16..23]; half 1: K = [8..15] and [24..31]
  union { bf16x16 v; uint4 q[2]; } u;
  u.q[0] = *reinterpret_cast<const uint4*>(p);
  u.q[1] = *reinterpret_cast<const uint4*>(p + 16);
  return u.v;
}

__device__ __forceinline__ f32x8 wmma_bf16(bf16x16 a, bf16x16 b, f32x8 c) {
  return __builtin_amdgcn_wmma_f32_16x16x32_bf16(false, a, false, b, (short)0, c,
                                                 false, false);
}

// ---------------- atomic helpers ----------------

__device__ __forceinline__ void atomicMaxF(float* addr, float val) {
  if (val >= 0.0f)
    atomicMax(reinterpret_cast<int*>(addr), __float_as_int(val));
  else
    atomicMin(reinterpret_cast<unsigned int*>(addr), __float_as_uint(val));
}

__device__ __forceinline__ void atomicAddF(float* addr, float val) {
  unsafeAtomicAdd(addr, val);   // hardware global_atomic_add_f32
}

// ---------------- elementwise / prep kernels ----------------

__global__ void k_fill(float* p, float v, int n) {
  int i = blockIdx.x * blockDim.x + threadIdx.x;
  if (i < n) p[i] = v;
}

__global__ void k_f2bf(const float* src, bf16* dst, int n) {
  int i = blockIdx.x * blockDim.x + threadIdx.x;
  if (i < n) dst[i] = (bf16)src[i];
}

// W: [K,N] row-major (f32) -> WT: [N,Kp] row-major (bf16), zero pad K..Kp
__global__ void k_wt(const float* W, bf16* WT, int K, int N, int Kp) {
  int i = blockIdx.x * blockDim.x + threadIdx.x;
  if (i >= N * Kp) return;
  int n = i / Kp, k = i % Kp;
  WT[i] = (k < K) ? (bf16)W[k * N + n] : (bf16)0.0f;
}

// ---------------- generic bf16 WMMA GEMM ----------------
// C[M,N] = A[M,K](bf16) * B[K,N] (given as BT[N,K] bf16), + optional bias,
// act: 0=none, 1=relu. Writes f32 C and/or bf16 Cbf.
// One wave -> 32x16 tile (two 16x16 accumulators sharing the B fragment).
__global__ void k_gemm(const bf16* __restrict__ A, const bf16* __restrict__ BT,
                       const float* __restrict__ bias,
                       float* __restrict__ C, bf16* __restrict__ Cbf,
                       int M, int N, int K, int act) {
  int wave = threadIdx.x >> 5;
  int lane = threadIdx.x & 31;
  int half = lane >> 4;
  int ln   = lane & 15;
  int m0 = blockIdx.x * 32;
  int n0 = (blockIdx.y * 8 + wave) * 16;
  if (n0 >= N) return;

  f32x8 acc0 = {};
  f32x8 acc1 = {};
  const bf16* arow0 = A + (m0 + ln) * K + half * 8;
  const bf16* arow1 = arow0 + 16 * K;
  const bf16* brow  = BT + (n0 + ln) * K + half * 8;
  for (int k0 = 0; k0 < K; k0 += 32) {
    bf16x16 b  = load_frag(brow + k0);
    bf16x16 a0 = load_frag(arow0 + k0);
    bf16x16 a1 = load_frag(arow1 + k0);
    acc0 = wmma_bf16(a0, b, acc0);
    acc1 = wmma_bf16(a1, b, acc1);
  }
  int n = n0 + ln;
  float bv = bias ? bias[n] : 0.0f;
#pragma unroll
  for (int r = 0; r < 8; ++r) {
    int m = m0 + r + half * 8;
    float v0 = acc0[r] + bv;
    float v1 = acc1[r] + bv;
    if (act == 1) {
      v0 = v0 > 0.0f ? v0 : 0.0f;
      v1 = v1 > 0.0f ? v1 : 0.0f;
    }
    if (C) {
      C[m * N + n]        = v0;
      C[(m + 16) * N + n] = v1;
    }
    if (Cbf) {
      Cbf[m * N + n]        = (bf16)v0;
      Cbf[(m + 16) * N + n] = (bf16)v1;
    }
  }
}

// ---------------- GAT attention pieces ----------------

// asrc[n,h] = sum_c h[n,h,c]*a_src[h,c]; likewise adst
__global__ void k_attn(const float* __restrict__ h, const float* __restrict__ a_src,
                       const float* __restrict__ a_dst,
                       float* __restrict__ asrc, float* __restrict__ adst,
                       int Nn, int Hh, int C) {
  int idx = blockIdx.x * blockDim.x + threadIdx.x;
  if (idx >= Nn * Hh) return;
  int n = idx / Hh, hh = idx % Hh;
  const float* hp = h + (n * Hh + hh) * C;
  const float* as = a_src + hh * C;
  const float* ad = a_dst + hh * C;
  float s1 = 0.0f, s2 = 0.0f;
  for (int c = 0; c < C; ++c) { float v = hp[c]; s1 += v * as[c]; s2 += v * ad[c]; }
  asrc[idx] = s1; adst[idx] = s2;
}

__global__ void k_edge_max(const float* __restrict__ asrc, const float* __restrict__ adst,
                           const int* __restrict__ src, const int* __restrict__ dst,
                           float* __restrict__ amax, int E, int Nn, int Hh) {
  int e = blockIdx.x * blockDim.x + threadIdx.x;
  if (e >= E + Nn) return;
  int s = e < E ? src[e] : e - E;
  int d = e < E ? dst[e] : e - E;
  for (int hh = 0; hh < Hh; ++hh) {
    float al = asrc[s * Hh + hh] + adst[d * Hh + hh];
    al = al > 0.0f ? al : 0.2f * al;   // leaky_relu 0.2
    atomicMaxF(&amax[d * Hh + hh], al);
  }
}

__global__ void k_edge_exp(const float* __restrict__ asrc, const float* __restrict__ adst,
                           const int* __restrict__ src, const int* __restrict__ dst,
                           const float* __restrict__ amax,
                           float* __restrict__ exbuf, float* __restrict__ den,
                           int E, int Nn, int Hh) {
  int e = blockIdx.x * blockDim.x + threadIdx.x;
  if (e >= E + Nn) return;
  int s = e < E ? src[e] : e - E;
  int d = e < E ? dst[e] : e - E;
  for (int hh = 0; hh < Hh; ++hh) {
    float al = asrc[s * Hh + hh] + adst[d * Hh + hh];
    al = al > 0.0f ? al : 0.2f * al;
    float ex = expf(al - amax[d * Hh + hh]);
    exbuf[e * Hh + hh] = ex;
    atomicAddF(&den[d * Hh + hh], ex);
  }
}

// out[d, pos] += h[s, pos] * ex[e,h]/den[d,h];  HC = Hh*128 (power of 2), shift = log2(HC)
__global__ void k_agg(const float* __restrict__ h, const float* __restrict__ exbuf,
                      const float* __restrict__ den,
                      const int* __restrict__ src, const int* __restrict__ dst,
                      float* __restrict__ out, int E, int Nn, int Hh, int shift) {
  int HC = 1 << shift;
  int total = (E + Nn) << shift;
  int idx = blockIdx.x * blockDim.x + threadIdx.x;
  if (idx >= total) return;
  int e   = idx >> shift;
  int pos = idx & (HC - 1);
  int hh  = pos >> 7;                 // C = 128 for both layers
  int s = e < E ? src[e] : e - E;
  int d = e < E ? dst[e] : e - E;
  float w = exbuf[e * Hh + hh] / den[d * Hh + hh];
  atomicAddF(&out[d * HC + pos], h[s * HC + pos] * w);
}

// buf = elu(buf + bias[f]); also bf16 copy.  F is power of two (mask = F-1)
__global__ void k_bias_elu(float* __restrict__ buf, const float* __restrict__ bias,
                           bf16* __restrict__ obf, int n, int mask) {
  int i = blockIdx.x * blockDim.x + threadIdx.x;
  if (i >= n) return;
  float v = buf[i] + bias[i & mask];
  v = v > 0.0f ? v : expm1f(v);
  buf[i] = v;
  if (obf) obf[i] = (bf16)v;
}

// ---------------- fused edge MLP ----------------
// Block = 256 thr (8 waves) handles 16 edges.
// A-tile [16 x 288] bf16 in LDS = [xo[src] | xo[dst] | edge_attr | 0-pad].
// GEMM1: relu(A @ We1 + be1) -> T [16 x 128] bf16 in LDS (wave w does cols w*16..).
// GEMM2: wave 0: out[16 x 16] = T @ We2 + be2 -> d_out.
__global__ void k_edge_mlp(const bf16* __restrict__ xo_bf, const float* __restrict__ eattr,
                           const int* __restrict__ src, const int* __restrict__ dst,
                           const bf16* __restrict__ We1t, const float* __restrict__ be1,
                           const bf16* __restrict__ We2t, const float* __restrict__ be2,
                           float* __restrict__ out2) {
  __shared__ __align__(16) bf16 Atile[16 * KP1];
  __shared__ __align__(16) bf16 Ttile[16 * 128];
  int e0 = blockIdx.x * 16;
  int t  = threadIdx.x;

  // stage A-tile: 16 rows x 9 chunks of 32 bf16 (64B per chunk, uint4 copies)
  if (t < 144) {
    int r = t / 9, seg = t % 9;
    int e = e0 + r;
    uint4* dp = reinterpret_cast<uint4*>(Atile + r * KP1 + seg * 32);
    if (seg < 8) {
      const bf16* spb = (seg < 4) ? (xo_bf + src[e] * DD + seg * 32)
                                  : (xo_bf + dst[e] * DD + (seg - 4) * 32);
      const uint4* sp = reinterpret_cast<const uint4*>(spb);
      dp[0] = sp[0]; dp[1] = sp[1]; dp[2] = sp[2]; dp[3] = sp[3];
    } else {
      const float4* ea = reinterpret_cast<const float4*>(eattr + e * EAA);
      float4 f0 = ea[0], f1 = ea[1], f2 = ea[2], f3 = ea[3];
      union { bf16 b[16]; uint4 q[2]; } u;
      u.b[0]  = (bf16)f0.x; u.b[1]  = (bf16)f0.y; u.b[2]  = (bf16)f0.z; u.b[3]  = (bf16)f0.w;
      u.b[4]  = (bf16)f1.x; u.b[5]  = (bf16)f1.y; u.b[6]  = (bf16)f1.z; u.b[7]  = (bf16)f1.w;
      u.b[8]  = (bf16)f2.x; u.b[9]  = (bf16)f2.y; u.b[10] = (bf16)f2.z; u.b[11] = (bf16)f2.w;
      u.b[12] = (bf16)f3.x; u.b[13] = (bf16)f3.y; u.b[14] = (bf16)f3.z; u.b[15] = (bf16)f3.w;
      uint4 z = make_uint4(0u, 0u, 0u, 0u);
      dp[0] = u.q[0]; dp[1] = u.q[1]; dp[2] = z; dp[3] = z;
    }
  }
  __syncthreads();

  int wave = t >> 5, lane = t & 31, half = lane >> 4, ln = lane & 15;

  { // hidden GEMM: wave -> cols [wave*16, wave*16+16)
    int n0 = wave * 16;
    f32x8 acc = {};
    const bf16* arow = Atile + ln * KP1 + half * 8;
    const bf16* brow = We1t + (n0 + ln) * KP1 + half * 8;
#pragma unroll
    for (int k0 = 0; k0 < KP1; k0 += 32) {
      bf16x16 a = load_frag(arow + k0);
      bf16x16 b = load_frag(brow + k0);
      acc = wmma_bf16(a, b, acc);
    }
    float bv = be1[n0 + ln];
#pragma unroll
    for (int r = 0; r < 8; ++r) {
      float v = acc[r] + bv;
      v = v > 0.0f ? v : 0.0f;
      Ttile[(r + half * 8) * 128 + n0 + ln] = (bf16)v;
    }
  }
  __syncthreads();

  if (wave == 0) { // projection GEMM: [16x128] @ [128x16]
    f32x8 acc = {};
    const bf16* arow = Ttile + ln * 128 + half * 8;
    const bf16* brow = We2t + ln * 128 + half * 8;
#pragma unroll
    for (int k0 = 0; k0 < 128; k0 += 32) {
      bf16x16 a = load_frag(arow + k0);
      bf16x16 b = load_frag(brow + k0);
      acc = wmma_bf16(a, b, acc);
    }
    float bv = be2[ln];
#pragma unroll
    for (int r = 0; r < 8; ++r)
      out2[(e0 + r + half * 8) * EAA + ln] = acc[r] + bv;
  }
}

// ---------------- host orchestration ----------------

extern "C" void kernel_launch(void* const* d_in, const int* in_sizes, int n_in,
                              void* d_out, int out_size, void* d_ws, size_t ws_size,
                              hipStream_t stream) {
  const float* x     = (const float*)d_in[0];
  const int*   ei    = (const int*)d_in[1];
  const float* eattr = (const float*)d_in[2];
  const float* W1    = (const float*)d_in[3];
  const float* as1   = (const float*)d_in[4];
  const float* ad1   = (const float*)d_in[5];
  const float* b1    = (const float*)d_in[6];
  const float* W2    = (const float*)d_in[7];
  const float* as2   = (const float*)d_in[8];
  const float* ad2   = (const float*)d_in[9];
  const float* b2    = (const float*)d_in[10];
  const float* Wo1   = (const float*)d_in[11];
  const float* bo1   = (const float*)d_in[12];
  const float* Wo2   = (const float*)d_in[13];
  const float* bo2   = (const float*)d_in[14];
  const float* We1   = (const float*)d_in[15];
  const float* be1   = (const float*)d_in[16];
  const float* We2   = (const float*)d_in[17];
  const float* be2   = (const float*)d_in[18];
  const int* srcI = ei;
  const int* dstI = ei + EE;

  char* ws = (char*)d_ws;
  size_t off = 0;
  auto alloc = [&](size_t bytes) -> void* {
    void* p = ws + off;
    off = (off + bytes + 255) & ~(size_t)255;
    return p;
  };
  bf16*  x_bf   = (bf16*)alloc((size_t)NN * 128 * 2);
  bf16*  W1t    = (bf16*)alloc(512 * 128 * 2);
  bf16*  W2t    = (bf16*)alloc(128 * 512 * 2);
  bf16*  Wo1t   = (bf16*)alloc(128 * 128 * 2);
  bf16*  Wo2t   = (bf16*)alloc(128 * 128 * 2);
  bf16*  We1t   = (bf16*)alloc(128 * KP1 * 2);
  bf16*  We2t   = (bf16*)alloc(16 * 128 * 2);
  float* h1     = (float*)alloc((size_t)NN * 512 * 4);
  float* asrc1  = (float*)alloc((size_t)NN * 4 * 4);
  float* adst1  = (float*)alloc((size_t)NN * 4 * 4);
  float* amax1  = (float*)alloc((size_t)NN * 4 * 4);
  float* den1   = (float*)alloc((size_t)NN * 4 * 4);
  float* ex1    = (float*)alloc((size_t)ETOT * 4 * 4);
  float* out1   = (float*)alloc((size_t)NN * 512 * 4);
  bf16*  h1a_bf = (bf16*)alloc((size_t)NN * 512 * 2);
  float* t2     = (float*)alloc((size_t)NN * 128 * 4);
  float* asrc2  = (float*)alloc((size_t)NN * 4);
  float* adst2  = (float*)alloc((size_t)NN * 4);
  float* amax2  = (float*)alloc((size_t)NN * 4);
  float* den2   = (float*)alloc((size_t)NN * 4);
  float* ex2    = (float*)alloc((size_t)ETOT * 4);
  float* out2n  = (float*)alloc((size_t)NN * 128 * 4);
  bf16*  h2_bf  = (bf16*)alloc((size_t)NN * 128 * 2);
  bf16*  m1_bf  = (bf16*)alloc((size_t)NN * 128 * 2);
  bf16*  xo_bf  = (bf16*)alloc((size_t)NN * 128 * 2);

  float* xo_out   = (float*)d_out;
  float* edge_out = (float*)d_out + (size_t)NN * 128;

  const int TB = 256;
  auto nb = [&](long n) { return (int)((n + TB - 1) / TB); };

  // --- weight / input prep ---
  k_f2bf<<<nb((long)NN * 128), TB, 0, stream>>>(x, x_bf, NN * 128);
  k_wt<<<nb(512 * 128), TB, 0, stream>>>(W1, W1t, 128, 512, 128);
  k_wt<<<nb(128 * 512), TB, 0, stream>>>(W2, W2t, 512, 128, 512);
  k_wt<<<nb(128 * 128), TB, 0, stream>>>(Wo1, Wo1t, 128, 128, 128);
  k_wt<<<nb(128 * 128), TB, 0, stream>>>(Wo2, Wo2t, 128, 128, 128);
  k_wt<<<nb(128 * KP1), TB, 0, stream>>>(We1, We1t, 272, 128, KP1);
  k_wt<<<nb(16 * 128), TB, 0, stream>>>(We2, We2t, 128, 16, 128);

  // --- GAT layer 1 ---
  k_gemm<<<dim3(NN / 32, 4), TB, 0, stream>>>(x_bf, W1t, nullptr, h1, nullptr,
                                              NN, 512, 128, 0);
  k_attn<<<nb(NN * 4), TB, 0, stream>>>(h1, as1, ad1, asrc1, adst1, NN, 4, 128);
  k_fill<<<nb(NN * 4), TB, 0, stream>>>(amax1, -INFINITY, NN * 4);
  k_fill<<<nb(NN * 4), TB, 0, stream>>>(den1, 0.0f, NN * 4);
  k_fill<<<nb((long)NN * 512), TB, 0, stream>>>(out1, 0.0f, NN * 512);
  k_edge_max<<<nb(ETOT), TB, 0, stream>>>(asrc1, adst1, srcI, dstI, amax1, EE, NN, 4);
  k_edge_exp<<<nb(ETOT), TB, 0, stream>>>(asrc1, adst1, srcI, dstI, amax1, ex1, den1,
                                          EE, NN, 4);
  k_agg<<<nb((long)ETOT * 512), TB, 0, stream>>>(h1, ex1, den1, srcI, dstI, out1,
                                                 EE, NN, 4, 9);
  k_bias_elu<<<nb((long)NN * 512), TB, 0, stream>>>(out1, b1, h1a_bf, NN * 512, 511);

  // --- GAT layer 2 (1 head) ---
  k_gemm<<<dim3(NN / 32, 1), TB, 0, stream>>>(h1a_bf, W2t, nullptr, t2, nullptr,
                                              NN, 128, 512, 0);
  k_attn<<<nb(NN), TB, 0, stream>>>(t2, as2, ad2, asrc2, adst2, NN, 1, 128);
  k_fill<<<nb(NN), TB, 0, stream>>>(amax2, -INFINITY, NN);
  k_fill<<<nb(NN), TB, 0, stream>>>(den2, 0.0f, NN);
  k_fill<<<nb((long)NN * 128), TB, 0, stream>>>(out2n, 0.0f, NN * 128);
  k_edge_max<<<nb(ETOT), TB, 0, stream>>>(asrc2, adst2, srcI, dstI, amax2, EE, NN, 1);
  k_edge_exp<<<nb(ETOT), TB, 0, stream>>>(asrc2, adst2, srcI, dstI, amax2, ex2, den2,
                                          EE, NN, 1);
  k_agg<<<nb((long)ETOT * 128), TB, 0, stream>>>(t2, ex2, den2, srcI, dstI, out2n,
                                                 EE, NN, 1, 7);
  k_bias_elu<<<nb((long)NN * 128), TB, 0, stream>>>(out2n, b2, h2_bf, NN * 128, 127);

  // --- node output MLP: xo = relu(h2@Wo1+bo1)@Wo2 + bo2 ---
  k_gemm<<<dim3(NN / 32, 1), TB, 0, stream>>>(h2_bf, Wo1t, bo1, nullptr, m1_bf,
                                              NN, 128, 128, 1);
  k_gemm<<<dim3(NN / 32, 1), TB, 0, stream>>>(m1_bf, Wo2t, bo2, xo_out, xo_bf,
                                              NN, 128, 128, 0);

  // --- fused edge MLP ---
  k_edge_mlp<<<EE / 16, TB, 0, stream>>>(xo_bf, eattr, srcI, dstI, We1t, be1,
                                         We2t, be2, edge_out);

  (void)in_sizes; (void)n_in; (void)out_size; (void)ws_size;
}
